// HPCDEv1_20194936226061
// MI455X (gfx1250) — matrix operand order repaired and checked by reading, not compiled, not run on hardware
//
#include <hip/hip_runtime.h>
#include <hip/hip_bf16.h>
#include <math.h>

typedef __attribute__((ext_vector_type(16))) _Float16     v16h;
typedef __attribute__((ext_vector_type(8)))  float        v8f;
typedef __attribute__((ext_vector_type(4)))  unsigned int u32x4;
typedef __attribute__((ext_vector_type(8)))  int          i32x8;
typedef __attribute__((ext_vector_type(4)))  int          i32x4;

#ifndef __has_builtin
#define __has_builtin(x) 0
#endif
#if __has_builtin(__builtin_amdgcn_tensor_load_to_lds) && __has_builtin(__builtin_amdgcn_s_wait_tensorcnt)
#define HAVE_TDM 1
#else
#define HAVE_TDM 0
#endif

// ---------------- problem constants ----------------
#define BV     32
#define TV     256
#define DM     64          // D_MODEL
#define DN     128         // D_NCDE
#define HHV    64          // HH
#define INPUTD 65          // INPUT_DIM
#define NTY    22          // NUM_TYPES
#define NOUT   (DN * INPUTD)      // 8320
#define NTIL   (NOUT / 16)        // 520 N-tiles for Wout

// ---------------- workspace layout (bytes) ----------------
#define WS_WINF  0            // uint32[4*4*32*8]   = 16384 B
#define WS_WHF   16384        // uint32[2*4*32*8]   = 8192 B
#define WS_WOUTF 24576        // uint32[2*520*32*8] = 1064960 B
#define WS_X     1089536      // float [B*T*65]     = 2129920 B
#define WS_ZS    3219456      // float [B*T*128]    = 4194304 B

// ---------------- output layout (floats) ----------------
#define OUT_ENC  0
#define OUT_ODE  (BV*TV*DM)                 // 524288
#define OUT_TP   (OUT_ODE + BV*TV*NTY)      // 704512
#define OUT_TM   (OUT_TP  + BV*TV*NTY)      // 884736

// ============================================================
// Swizzle row-major f32 W[K][N] (K=ktiles*32, N=ntiles*16) into
// per-lane-contiguous f16 WMMA B fragments (contiguous K order, as in load_a_lds:
// lanes 0-15 K=0..15, lanes 16-31 K=16..31, 2 K packed per dword).
// ============================================================
__global__ void hpcde_swizzleB(const float* __restrict__ W,
                               unsigned int* __restrict__ F,
                               int ktiles, int ntiles, int N) {
    int idx = blockIdx.x * 256 + threadIdx.x;
    int total = ktiles * ntiles * 32 * 8;
    if (idx >= total) return;
    int v    = idx & 7;
    int lane = (idx >> 3) & 31;
    int frag = idx >> 8;
    int nt   = frag % ntiles;
    int kt   = frag / ntiles;
    int n  = nt * 16 + (lane & 15);
    int k  = kt * 32 + (lane >> 4) * 16 + 2 * v;
    _Float16 lo = (_Float16)W[(size_t)k * N + n];
    _Float16 hi = (_Float16)W[(size_t)(k + 1) * N + n];
    unsigned short lb, hb;
    __builtin_memcpy(&lb, &lo, 2);
    __builtin_memcpy(&hb, &hi, 2);
    F[idx] = (unsigned int)lb | ((unsigned int)hb << 16);
}

// ============================================================
// Build control path X[b][t][i]
// ============================================================
__global__ void hpcde_path(const int* __restrict__ etype,
                           const float* __restrict__ etime,
                           const float* __restrict__ emb,
                           float* __restrict__ X) {
    int idx = blockIdx.x * 256 + threadIdx.x;
    const int total = BV * TV * INPUTD;
    if (idx >= total) return;
    int i  = idx % INPUTD;
    int bt = idx / INPUTD;
    float tme = etime[bt];
    int   ty  = etype[bt];
    float msk = (ty != 0) ? 1.0f : 0.0f;
    float val;
    if (i == 0) {
        val = tme;
    } else {
        int d = i - 1;
        float pv = powf(10000.0f, (2.0f * (float)(d / 2)) / (float)DM);
        float r  = tme / pv;
        float te = ((d & 1) == 0) ? sinf(r) : cosf(r);
        val = emb[(size_t)ty * DM + d] + te * msk;
    }
    X[idx] = val;
}

// ---- A fragment (16x32 f16) from an LDS f32 matrix [16][ldk] ----
__device__ __forceinline__ v16h load_a_lds(const float* Zm, int ldk, int kt, int lane) {
    int m  = lane & 15;
    int kh = lane >> 4;                 // K half: +0 / +16
    const float* row = Zm + (size_t)m * ldk + kt * 32 + kh * 16;
    v16h a;
#pragma unroll
    for (int v = 0; v < 8; ++v) {
        int kb = 2 * v;
        a[2 * v]     = (_Float16)row[kb];
        a[2 * v + 1] = (_Float16)row[kb + 1];
    }
    return a;
}

// ---- B fragment: two contiguous b128 loads per lane ----
__device__ __forceinline__ v16h load_b_frag(const unsigned int* __restrict__ F,
                                            int ntiles, int kt, int nt, int lane) {
    const unsigned int* p = F + (((size_t)kt * ntiles + nt) * 32 + lane) * 8;
    return *(const v16h*)p;
}

#define WMMA_F16(a, b, c) \
    __builtin_amdgcn_wmma_f32_16x16x32_f16(false, (a), false, (b), (short)0, (c), false, false)

#if HAVE_TDM
// ---- TDM: async 2D tile load (16 rows x 130 f32, row stride T*65) ----
__device__ __forceinline__ void tdm_load_x_tile(const float* gsrc, unsigned lds_off) {
    unsigned long long ga = (unsigned long long)(size_t)gsrc;
    u32x4 g0;
    g0[0] = 1u;                                  // count=1, user mode, no gather
    g0[1] = lds_off;                             // lds_addr (bytes)
    g0[2] = (unsigned)(ga & 0xFFFFFFFFu);        // global_addr[31:0]
    g0[3] = (unsigned)((ga >> 32) & 0x01FFFFFFu) // global_addr[56:32]
          | (2u << 30);                          // type=2 ("image")
    const unsigned td0 = 130u, td1 = 16u;
    const unsigned long long st0 = (unsigned long long)(TV * INPUTD); // 16640
    i32x8 g1;
    g1[0] = (int)(2u << 16);                               // data_size=4B
    g1[1] = (int)((td0 & 0xFFFFu) << 16);                  // tensor_dim0[15:0]
    g1[2] = (int)((td0 >> 16) | ((td1 & 0xFFFFu) << 16));  // td0 hi | td1 lo
    g1[3] = (int)((td1 >> 16) | (130u << 16));             // td1 hi | tile_dim0=130
    g1[4] = (int)(16u);                                    // tile_dim1=16, tile_dim2=0
    g1[5] = (int)(st0 & 0xFFFFFFFFu);                      // tensor_dim0_stride lo
    g1[6] = (int)((st0 >> 32) & 0xFFFFu);                  // stride hi | dim1_stride lo
    g1[7] = 0;
    i32x4 z4 = {0, 0, 0, 0};
    i32x8 z8 = {0, 0, 0, 0, 0, 0, 0, 0};
    // 6-arg variant (this toolchain): (g0, g1, g2, g3, g4, cpol)
    __builtin_amdgcn_tensor_load_to_lds(g0, g1, z4, z4, z8, 0);
}
#endif

// ============================================================
// Persistent RK4 scan.  grid = 2 blocks x 256 threads (8 waves).
// Block bb owns batch rows [bb*16, bb*16+16) -> one WMMA M-tile.
// ============================================================
__global__ void __launch_bounds__(256, 1)
hpcde_scan(const float* __restrict__ X,
           const unsigned int* __restrict__ WinF,
           const unsigned int* __restrict__ WhF,
           const unsigned int* __restrict__ WoutF,
           const float* __restrict__ initW, const float* __restrict__ initb,
           const float* __restrict__ binv,  const float* __restrict__ bhv,
           const float* __restrict__ boutv,
           const float* __restrict__ odeW,  const float* __restrict__ odeB,
           const float* __restrict__ betap,
           float* __restrict__ zs, float* __restrict__ ode_out) {
    __shared__ float Zb [16][DN];
    __shared__ float Zst[16][DN];
    __shared__ float Ks [16][DN];
    __shared__ float DZ [16][DN];
    __shared__ float H1 [16][HHV];
    __shared__ float H2 [16][HHV];
    __shared__ float DX [16][INPUTD];
    __shared__ float XT [16][2 * INPUTD];   // TDM-staged X tile (rows t, t+1)
    __shared__ float Aac[16][NTY];
    __shared__ float Asm[16][NTY];
    __shared__ float KA [16][NTY];

    const int tid  = threadIdx.x;
    const int wave = tid >> 5;
    const int lane = tid & 31;
    const int m0   = blockIdx.x * 16;
    const float beta = betap[0];

#if HAVE_TDM
    const unsigned xt_off = (unsigned)(size_t)(void*)&XT[0][0];  // LDS byte offset
    if (wave == 0)
        tdm_load_x_tile(X + ((size_t)m0 * TV + 0) * INPUTD, xt_off);
#endif

    // ---- z0 = X[:,0,:] @ init_W + init_b ; a0 = 0 ----
    for (int idx = tid; idx < 16 * DN; idx += 256) {
        int m = idx / DN, h = idx % DN;
        const float* x0 = &X[(size_t)(m0 + m) * TV * INPUTD];
        float s = initb[h];
        for (int i = 0; i < INPUTD; ++i) s += x0[i] * initW[(size_t)i * DN + h];
        Zb[m][h] = s;
        zs[((size_t)(m0 + m) * TV + 0) * DN + h] = s;
    }
    for (int idx = tid; idx < 16 * NTY; idx += 256) {
        int m = idx / NTY, j = idx % NTY;
        Aac[m][j] = 0.0f;
        ode_out[((size_t)(m0 + m) * TV + 0) * NTY + j] = 0.0f;
    }
    __syncthreads();

    for (int t = 0; t < TV - 1; ++t) {
        // reset stage state (independent of X tile)
        for (int idx = tid; idx < 16 * DN; idx += 256) {
            int m = idx / DN, h = idx % DN;
            Zst[m][h] = Zb[m][h];
            Ks[m][h]  = 0.0f;
        }
        for (int idx = tid; idx < 16 * NTY; idx += 256) Asm[idx / NTY][idx % NTY] = 0.0f;

#if HAVE_TDM
        if (wave == 0) __builtin_amdgcn_s_wait_tensorcnt(0);
        __syncthreads();                          // XT (rows t,t+1) visible
        for (int idx = tid; idx < 16 * INPUTD; idx += 256) {
            int m = idx / INPUTD, i = idx % INPUTD;
            DX[m][i] = XT[m][INPUTD + i] - XT[m][i];
        }
        __syncthreads();                          // DX built; XT dead
        if (wave == 0 && t + 1 < TV - 1)          // overlap next tile with stages
            tdm_load_x_tile(X + ((size_t)m0 * TV + (t + 1)) * INPUTD, xt_off);
#else
        for (int idx = tid; idx < 16 * INPUTD; idx += 256) {
            int m = idx / INPUTD, i = idx % INPUTD;
            const float* xb = &X[((size_t)(m0 + m) * TV + t) * INPUTD];
            DX[m][i] = xb[INPUTD + i] - xb[i];
        }
        __syncthreads();
#endif

        for (int s = 0; s < 4; ++s) {
            for (int idx = tid; idx < 16 * DN; idx += 256) DZ[idx / DN][idx % DN] = 0.0f;
            __syncthreads();

            // ---- h1 = relu(Zst @ Win + bin): waves 0..3 ----
            if (wave < 4) {
                const int nt = wave;
                v8f c = {};
#pragma unroll
                for (int kt = 0; kt < 4; ++kt) {
                    v16h a = load_a_lds(&Zst[0][0], DN, kt, lane);
                    v16h b = load_b_frag(WinF, 4, kt, nt, lane);
                    c = WMMA_F16(a, b, c);
                }
                int n  = nt * 16 + (lane & 15);
                int mh = lane >> 4;
                float bb = binv[n];
#pragma unroll
                for (int v = 0; v < 8; ++v) {
                    float val = c[v] + bb;
                    H1[v + mh * 8][n] = val > 0.0f ? val : 0.0f;
                }
            } else {
                // ---- intensity head, concurrently on waves 4..7 ----
                for (int idx = (wave - 4) * 32 + lane; idx < 16 * NTY; idx += 128) {
                    int m = idx / NTY, j = idx % NTY;
                    float sum = odeB[j];
                    for (int k = 0; k < DN; ++k) sum += Zst[m][k] * odeW[(size_t)k * NTY + j];
                    float xb = beta * sum;
                    float sp = (xb > 20.0f) ? xb : log1pf(expf(xb));
                    KA[m][j] = sp / beta;
                }
            }
            __syncthreads();

            // ---- h2 = relu(H1 @ Wh + bh): waves 0..3 ----
            if (wave < 4) {
                const int nt = wave;
                v8f c = {};
#pragma unroll
                for (int kt = 0; kt < 2; ++kt) {
                    v16h a = load_a_lds(&H1[0][0], HHV, kt, lane);
                    v16h b = load_b_frag(WhF, 4, kt, nt, lane);
                    c = WMMA_F16(a, b, c);
                }
                int n  = nt * 16 + (lane & 15);
                int mh = lane >> 4;
                float bb = bhv[n];
#pragma unroll
                for (int v = 0; v < 8; ++v) {
                    float val = c[v] + bb;
                    H2[v + mh * 8][n] = val > 0.0f ? val : 0.0f;
                }
            }
            __syncthreads();

            // ---- G = tanh(H2 @ Wout + bout); dz += G . dx ----
            // A fragments hoisted: loop-invariant across all 65 tiles.
            {
                const v16h a0 = load_a_lds(&H2[0][0], HHV, 0, lane);
                const v16h a1 = load_a_lds(&H2[0][0], HHV, 1, lane);
                const int mh  = lane >> 4;
                for (int nt = wave; nt < NTIL; nt += 8) {
                    if (nt + 8 < NTIL) {
                        __builtin_prefetch(WoutF + (((size_t)0 * NTIL + (nt + 8)) * 32 + lane) * 8, 0, 3);
                        __builtin_prefetch(WoutF + (((size_t)1 * NTIL + (nt + 8)) * 32 + lane) * 8, 0, 3);
                    }
                    v16h b0 = load_b_frag(WoutF, NTIL, 0, nt, lane);
                    v16h b1 = load_b_frag(WoutF, NTIL, 1, nt, lane);
                    v8f c = {};
                    c = WMMA_F16(a0, b0, c);
                    c = WMMA_F16(a1, b1, c);

                    const int nbase = nt * 16;
                    const int n  = nbase + (lane & 15);
                    const int h  = n / INPUTD;
                    const int i  = n % INPUTD;
                    const int hA = nbase / INPUTD;          // first h in tile
                    const int hB = (nbase + 15) / INPUTD;   // last h in tile (hA or hA+1)
                    const float bo = boutv[n];
#pragma unroll
                    for (int v = 0; v < 8; ++v) {
                        const int m = v + mh * 8;           // same for all 16 lanes in half
                        float g   = tanhf(c[v] + bo);
                        float val = g * DX[m][i];
                        // segmented reduction over the 16-lane half (<=2 h groups)
                        float sA = (h == hA) ? val : 0.0f;
                        float sB = (h == hA) ? 0.0f : val;
#pragma unroll
                        for (int off = 1; off < 16; off <<= 1) {
                            sA += __shfl_xor(sA, off, 16);
                            sB += __shfl_xor(sB, off, 16);
                        }
                        if ((lane & 15) == 0) {
                            atomicAdd(&DZ[m][hA], sA);
                            if (hB != hA) atomicAdd(&DZ[m][hB], sB);
                        }
                    }
                }
            }
            __syncthreads();

            // ---- RK4 combine ----
            const float wc = (s == 0 || s == 3) ? (1.0f / 6.0f) : (2.0f / 6.0f);
            const float cc = (s <= 1) ? 0.5f : 1.0f;
            for (int idx = tid; idx < 16 * DN; idx += 256) {
                int m = idx / DN, h = idx % DN;
                float k = DZ[m][h];
                Ks[m][h] += wc * k;
                if (s < 3) Zst[m][h] = Zb[m][h] + cc * k;
            }
            for (int idx = tid; idx < 16 * NTY; idx += 256)
                Asm[idx / NTY][idx % NTY] += wc * KA[idx / NTY][idx % NTY];
            __syncthreads();
        }

        // ---- advance state, emit zs[:,t+1], ode_out[:,t+1] ----
        for (int idx = tid; idx < 16 * DN; idx += 256) {
            int m = idx / DN, h = idx % DN;
            float zn = Zb[m][h] + Ks[m][h];
            Zb[m][h] = zn;
            zs[((size_t)(m0 + m) * TV + (t + 1)) * DN + h] = zn;
        }
        for (int idx = tid; idx < 16 * NTY; idx += 256) {
            int m = idx / NTY, j = idx % NTY;
            float an = Aac[m][j] + Asm[m][j];
            Aac[m][j] = an;
            ode_out[((size_t)(m0 + m) * TV + (t + 1)) * NTY + j] = an;
        }
        __syncthreads();
    }
}

// ============================================================
// Epilogue heads.  One 64-thread block per (b,t).
// ============================================================
__global__ void hpcde_readout(const float* __restrict__ zs,
                              const float* __restrict__ mask,
                              const float* __restrict__ rW, const float* __restrict__ rb,
                              const float* __restrict__ W1, const float* __restrict__ W2,
                              const float* __restrict__ tW,
                              float* __restrict__ out_enc,
                              float* __restrict__ out_tp,
                              float* __restrict__ out_tm) {
    __shared__ float enc[DM];
    __shared__ float ve[DM];
    const int bt = blockIdx.x;
    const int d  = threadIdx.x;
    const float* z = &zs[(size_t)bt * DN];
    float mk = mask[bt];
    float s = rb[d];
    for (int k = 0; k < DN; ++k) s += z[k] * rW[(size_t)k * DM + d];
    s *= mk;
    enc[d] = s;
    out_enc[(size_t)bt * DM + d] = s;
    __syncthreads();
    float u = 0.0f;
    for (int k = 0; k < DM; ++k) u += enc[k] * W1[(size_t)k * DM + d];
    ve[d] = (u > 0.0f) ? u : (expf(u) - 1.0f);
    __syncthreads();
    if (d < NTY) {
        float tp = 0.0f;
        for (int k = 0; k < DM; ++k) tp += ve[k] * W2[(size_t)k * NTY + d];
        out_tp[(size_t)bt * NTY + d] = tp * mk;
    }
    if (d == 0) {
        float tm = 0.0f;
        for (int k = 0; k < DM; ++k) tm += enc[k] * tW[k];
        out_tm[bt] = tm * mk;
    }
}

// ============================================================
extern "C" void kernel_launch(void* const* d_in, const int* in_sizes, int n_in,
                              void* d_out, int out_size, void* d_ws, size_t ws_size,
                              hipStream_t stream) {
    (void)in_sizes; (void)n_in; (void)out_size; (void)ws_size;
    const int*   etype = (const int*)  d_in[0];
    const float* etime = (const float*)d_in[1];
    const float* rmask = (const float*)d_in[2];
    const float* emb   = (const float*)d_in[3];
    const float* initW = (const float*)d_in[4];
    const float* initb = (const float*)d_in[5];
    const float* rW    = (const float*)d_in[6];
    const float* rb    = (const float*)d_in[7];
    const float* Win   = (const float*)d_in[8];
    const float* binv  = (const float*)d_in[9];
    const float* Wh    = (const float*)d_in[10];
    const float* bhv   = (const float*)d_in[11];
    const float* Wout  = (const float*)d_in[12];
    const float* boutv = (const float*)d_in[13];
    const float* odeW  = (const float*)d_in[14];
    const float* odeB  = (const float*)d_in[15];
    const float* betap = (const float*)d_in[16];
    const float* tpW1  = (const float*)d_in[17];
    const float* tpW2  = (const float*)d_in[18];
    const float* tW    = (const float*)d_in[19];

    char* ws = (char*)d_ws;
    unsigned int* WinF  = (unsigned int*)(ws + WS_WINF);
    unsigned int* WhF   = (unsigned int*)(ws + WS_WHF);
    unsigned int* WoutF = (unsigned int*)(ws + WS_WOUTF);
    float*        Xb    = (float*)(ws + WS_X);
    float*        zsb   = (float*)(ws + WS_ZS);
    float* out = (float*)d_out;

    hpcde_swizzleB<<<(4 * 4 * 32 * 8 + 255) / 256, 256, 0, stream>>>(Win, WinF, 4, 4, 64);
    hpcde_swizzleB<<<(2 * 4 * 32 * 8 + 255) / 256, 256, 0, stream>>>(Wh, WhF, 2, 4, 64);
    hpcde_swizzleB<<<(2 * NTIL * 32 * 8 + 255) / 256, 256, 0, stream>>>(Wout, WoutF, 2, NTIL, NOUT);

    hpcde_path<<<(BV * TV * INPUTD + 255) / 256, 256, 0, stream>>>(etype, etime, emb, Xb);

    hpcde_scan<<<2, 256, 0, stream>>>(Xb, WinF, WhF, WoutF,
                                      initW, initb, binv, bhv, boutv,
                                      odeW, odeB, betap,
                                      zsb, out + OUT_ODE);

    hpcde_readout<<<BV * TV, 64, 0, stream>>>(zsb, rmask, rW, rb, tpW1, tpW2, tW,
                                              out + OUT_ENC, out + OUT_TP, out + OUT_TM);
}
